// EgocentricHiveMindCritic_11957188952411
// MI455X (gfx1250) — matrix operand order repaired and checked by reading, not compiled, not run on hardware
//
#include <hip/hip_runtime.h>
#include <hip/hip_bf16.h>

typedef __attribute__((ext_vector_type(16))) _Float16 v16h;
typedef __attribute__((ext_vector_type(8)))  _Float16 v8h;
typedef __attribute__((ext_vector_type(8)))  float    v8f;
typedef __attribute__((ext_vector_type(4)))  int      v4i;
typedef _Float16 h16;

// ---------------------------------------------------------------------------
// CDNA5 async global->LDS copy (ASYNCcnt path), with portable fallback.
// Builtin signature (from clang diagnostic): (int4 addrspace(1)*,
// int4 addrspace(3)*, imm offset, imm cpol).
// ---------------------------------------------------------------------------
#if defined(__has_builtin)
#if __has_builtin(__builtin_amdgcn_global_load_async_to_lds_b128)
#define ASYNC_LDS 1
#endif
#endif
#ifndef ASYNC_LDS
#define ASYNC_LDS 0
#endif

#if ASYNC_LDS
typedef __attribute__((address_space(1))) v4i* gv4p;
typedef __attribute__((address_space(3))) v4i* lv4p;
#endif

__device__ __forceinline__ void lds_copy16(void* lds, const void* g) {
#if ASYNC_LDS
  __builtin_amdgcn_global_load_async_to_lds_b128(
      (gv4p)(unsigned long long)g,
      (lv4p)(unsigned)(unsigned long long)lds, 0, 0);
#else
  *(v8h*)lds = *(const v8h*)g;
#endif
}

__device__ __forceinline__ void lds_copy_wait() {
#if ASYNC_LDS
#if __has_builtin(__builtin_amdgcn_s_wait_asynccnt)
  __builtin_amdgcn_s_wait_asynccnt(0);
#else
  asm volatile("s_wait_asynccnt 0" ::: "memory");
#endif
#endif
}

// stage a rows x cols f16 panel (row-major, cols % 8 == 0) into LDS; barrier.
__device__ __forceinline__ void stage_rows(const h16* src, int ld_src,
                                           int rows, int cols, h16* dst) {
  int cpr = cols >> 3;                 // 16B chunks per row
  int chunks = rows * cpr;
  for (int i = threadIdx.x; i < chunks; i += 256) {
    int r = i / cpr, c = (i - r * cpr) << 3;
    lds_copy16(dst + r * cols + c, src + (size_t)r * ld_src + c);
  }
  lds_copy_wait();
  __syncthreads();
}

// ---------------------------------------------------------------------------
// WMMA helpers: ISA-exact register layouts (CDNA5 ISA §7.12.2)
// ---------------------------------------------------------------------------
__device__ __forceinline__ v16h load_a_tile(const h16* A, int lda) {
  int lane = threadIdx.x & 31;
  const h16* p = A + (size_t)(lane & 15) * lda + ((lane >> 4) << 3);
  v8h lo = *(const v8h*)p;
  v8h hi = *(const v8h*)(p + 16);
  v16h r;
#pragma unroll
  for (int i = 0; i < 8; ++i) { r[i] = lo[i]; r[i + 8] = hi[i]; }
  return r;
}

__device__ __forceinline__ v16h load_b_tile(const h16* W, int ldw) {
  int lane = threadIdx.x & 31;
  const h16* p = W + (size_t)(lane & 15) * ldw + ((lane >> 4) << 3);
  v8h lo = *(const v8h*)p;
  v8h hi = *(const v8h*)(p + 16);
  v16h r;
#pragma unroll
  for (int i = 0; i < 8; ++i) { r[i] = lo[i]; r[i + 8] = hi[i]; }
  return r;
}

__device__ __forceinline__ v8f wmma_kloop(const h16* A, int lda,
                                          const h16* W, int ldw, int K) {
  v8f acc = {0.f, 0.f, 0.f, 0.f, 0.f, 0.f, 0.f, 0.f};
  for (int k = 0; k < K; k += 32) {
    v16h a = load_a_tile(A + k, lda);
    v16h b = load_b_tile(W + k, ldw);
    acc = __builtin_amdgcn_wmma_f32_16x16x32_f16(false, a, false, b,
                                                 (short)0, acc, false, false);
  }
  return acc;
}

__device__ __forceinline__ float sigm(float x) { return 1.f / (1.f + __expf(-x)); }

#define MTOK 4096
#define SPAD 320
#define XROW 384
#define EPAD 64
#define EF   64

// ---------------------------------------------------------------------------
// Prep: pack/convert all weights to f16 with zero K-padding; init h/c
// ---------------------------------------------------------------------------
__global__ void prep_misc_kernel(
    const float* self_w, const float* ew0, const float* ew1, const float* ew2,
    const float* in_proj_w, const float* pool_w, const float* w_ih, const float* w_hh,
    const float* h0, const float* c0,
    h16* Wc, h16* wfeat, h16* inproj, h16* poolw, h16* wih, h16* whh,
    h16* hbuf, float* cbuf) {
  int idx = blockIdx.x * blockDim.x + threadIdx.x;
  if (idx < 256 * 320) {
    int r = idx / 320, k = idx - r * 320;
    float v = 0.f;
    if (k < 290) {
      if (r < 64)       v = self_w[r * 580 + k] + self_w[r * 580 + 290 + k];
      else if (r < 128) v = ew0[(r - 64) * 306 + 16 + k];
      else if (r < 192) v = ew1[(r - 128) * 314 + 24 + k];
      else              v = ew2[(r - 192) * 322 + 32 + k];
    }
    Wc[idx] = (h16)v; return;
  }
  idx -= 256 * 320;
  if (idx < 3 * 64 * 32) {
    int seg = idx / 2048, rem = idx - seg * 2048;
    int n = rem >> 5, k = rem & 31;
    float v = 0.f;
    if (seg == 0)      { if (k < 16) v = ew0[n * 306 + k]; }
    else if (seg == 1) { if (k < 24) v = ew1[n * 314 + k]; }
    else               {             v = ew2[n * 322 + k]; }
    wfeat[idx] = (h16)v; return;
  }
  idx -= 3 * 64 * 32;
  if (idx < 192 * 64) { inproj[idx] = (h16)in_proj_w[idx]; return; }
  idx -= 192 * 64;
  if (idx < 128 * 384) {
    int r = idx / 384, k = idx - r * 384;
    float v = 0.f;
    if (k < 290)       v = pool_w[r * 354 + k];
    else if (k >= 320) v = pool_w[r * 354 + 290 + (k - 320)];
    poolw[idx] = (h16)v; return;
  }
  idx -= 128 * 384;
  if (idx < 512 * 128) { wih[idx] = (h16)w_ih[idx]; return; }
  idx -= 512 * 128;
  if (idx < 512 * 128) { whh[idx] = (h16)w_hh[idx]; return; }
  idx -= 512 * 128;
  if (idx < 128 * 128) { hbuf[idx] = (h16)h0[idx]; cbuf[idx] = c0[idx]; return; }
}

__global__ void prep_comb_kernel(const float* sa_w, const float* sa_b,
                                 const float* ow, const float* ob,
                                 h16* wcomb, float* bcomb) {
  int o = blockIdx.x * blockDim.x + threadIdx.x;
  if (o >= 64 * 64) return;
  int n = o >> 6, j = o & 63;
  float acc = 0.f;
  for (int t = 0; t < 64; ++t) acc += sa_w[n * 64 + t] * ow[t * 64 + j];
  wcomb[n * 64 + j] = (h16)acc;
  if (j == 0) {
    float b = sa_b[n];
    for (int t = 0; t < 64; ++t) b += sa_w[n * 64 + t] * ob[t];
    bcomb[n] = b;
  }
}

__global__ void build_xself_kernel(const float* xa, const float* xl, const float* xs,
                                   const float* lw, const float* lb, h16* xpool) {
  int m = blockIdx.x;
  const float* xlm = xl + m * 32;
  for (int c = threadIdx.x; c < SPAD; c += 64) {
    float v;
    if (c < 16) v = xa[m * 16 + c];
    else if (c < 286) {
      int i = c - 16, f = i / 30, p = i - f * 30;
      v = fmaxf(lb[f] + lw[f * 3] * xlm[p] + lw[f * 3 + 1] * xlm[p + 1]
                      + lw[f * 3 + 2] * xlm[p + 2], 0.f);
    } else if (c < 290) v = xs[m * 4 + (c - 286)];
    else v = 0.f;
    xpool[(size_t)m * XROW + c] = (h16)v;
  }
}

__global__ void featpad_kernel(const float* f0, const float* f1, const float* f2, h16* fp) {
  int idx = blockIdx.x * blockDim.x + threadIdx.x;
  if (idx >= 3 * 65536 * 32) return;
  int seg = idx / (65536 * 32), rem = idx - seg * (65536 * 32);
  int row = rem >> 5, k = rem & 31;
  float v = 0.f;
  if (seg == 0)      { if (k < 16) v = f0[row * 16 + k]; }
  else if (seg == 1) { if (k < 24) v = f1[row * 24 + k]; }
  else               {             v = f2[row * 32 + k]; }
  fp[idx] = (h16)v;
}

// ---------------------------------------------------------------------------
// GEMMs: 8 waves/block, weight strip (16 x K) async-staged in LDS and shared
// across the block's 8 row tiles. grid = (N/16, Mtiles/8)
// ---------------------------------------------------------------------------
__global__ __launch_bounds__(256) void gemm_g0_kernel(const h16* A, const h16* W,
                                                      float* C) {
  __shared__ __align__(16) h16 Bs[16 * 320];
  int w = threadIdx.x >> 5, lane = threadIdx.x & 31;
  int row0 = (blockIdx.y * 8 + w) * 16, col0 = blockIdx.x * 16;
  stage_rows(W + (size_t)col0 * 320, 320, 16, 320, Bs);
  v8f d = wmma_kloop(A + (size_t)row0 * XROW, XROW, Bs, 320, 320);
  int n = col0 + (lane & 15), mb = row0 + ((lane >> 4) << 3);
#pragma unroll
  for (int r = 0; r < 8; ++r) C[(size_t)(mb + r) * 256 + n] = d[r];
}

__global__ void zself_kernel(const float* g0, const float* self_b, h16* zent) {
  int idx = blockIdx.x * blockDim.x + threadIdx.x;
  int m = idx >> 10, rem = idx & 1023, ep = rem >> 6, n = rem & 63;
  if (ep == 0)
    zent[((size_t)m * EPAD) * EF + n] =
        (h16)fmaxf(g0[(size_t)m * 256 + n] + self_b[n], 0.f);
  else
    zent[((size_t)m * EPAD + 48 + ep) * EF + n] = (h16)0.f;
}

__global__ __launch_bounds__(256) void gemm_feat_kernel(
    const h16* A, const h16* W, const float* g0, int base_off,
    const float* bias, h16* zent, int erow0) {
  __shared__ __align__(16) h16 Bs[16 * 32];
  int w = threadIdx.x >> 5, lane = threadIdx.x & 31;
  int row0 = (blockIdx.y * 8 + w) * 16, col0 = blockIdx.x * 16;
  stage_rows(W + (size_t)col0 * 32, 32, 16, 32, Bs);
  v8f d = wmma_kloop(A + (size_t)row0 * 32, 32, Bs, 32, 32);
  int n = col0 + (lane & 15), mb = row0 + ((lane >> 4) << 3);
  float b = bias[n];
#pragma unroll
  for (int r = 0; r < 8; ++r) {
    int rg = mb + r, m = rg >> 4, e = rg & 15;
    float v = fmaxf(d[r] + g0[(size_t)m * 256 + base_off + n] + b, 0.f);
    zent[((size_t)m * EPAD + erow0 + e) * EF + n] = (h16)v;
  }
}

// ---------------------------------------------------------------------------
// Fused attention per token: z_ent tile async-staged in LDS; qkv WMMA ->
// masked softmax -> combined (out_proj . sa_w) WMMA -> residual+relu ->
// masked mean pool
// ---------------------------------------------------------------------------
__global__ __launch_bounds__(256) void attn_kernel(
    const h16* zent, const h16* inproj, const float* inproj_b,
    const h16* wcomb, const float* bcomb,
    const float* mask0, const float* mask1, const float* mask2, h16* xpool) {
  __shared__ __align__(16) h16 zL[64 * 64];        // staged z_ent tile
  __shared__ __align__(16) h16 qkvL[64][192];
  __shared__ __align__(16) h16 attnL[64][64];
  __shared__ float maskL[64];
  __shared__ float zpoolL[64];
  int m = blockIdx.x, tid = threadIdx.x, w = tid >> 5, lane = tid & 31;

  if (tid < 64) {
    zpoolL[tid] = 0.f;
    float mv = 0.f;
    if (tid == 0) mv = 1.f;
    else if (tid <= 16) mv = mask0[m * 16 + tid - 1];
    else if (tid <= 32) mv = mask1[m * 16 + tid - 17];
    else if (tid <= 48) mv = mask2[m * 16 + tid - 33];
    maskL[tid] = mv;
  }
  for (int i = tid; i < 64 * 15; i += 256)          // zero pad rows 49..63
    attnL[49 + i / 64][i % 64] = (h16)0.f;
  stage_rows(zent + (size_t)m * EPAD * EF, EF, 64, 64, zL);  // includes barrier

  // qkv = z_ent @ in_proj.T + b  (64x192, 48 tiles, 6 per wave)
#pragma unroll
  for (int i = 0; i < 6; ++i) {
    int t = w * 6 + i, rt = t & 3, ct = t >> 2;
    v8f d = wmma_kloop(zL + rt * 16 * EF, EF, inproj + ct * 16 * EF, EF, EF);
    int n = ct * 16 + (lane & 15), mb = rt * 16 + ((lane >> 4) << 3);
    float b = inproj_b[n];
#pragma unroll
    for (int r = 0; r < 8; ++r) qkvL[mb + r][n] = (h16)(d[r] + b);
  }
  __syncthreads();

  // masked softmax attention: one thread per (query e<49, head h)
  if (tid < 196) {
    int e = tid % 49, h = tid / 49;
    const h16* q = &qkvL[e][h * 16];
    float mx = -1e30f;
    for (int j = 0; j < 49; ++j) {
      if (maskL[j] != 0.f) {
        const h16* kk = &qkvL[j][64 + h * 16];
        float s = 0.f;
#pragma unroll
        for (int d2 = 0; d2 < 16; ++d2) s += (float)q[d2] * (float)kk[d2];
        mx = fmaxf(mx, s * 0.25f);
      }
    }
    float acc[16]; float sum = 0.f;
#pragma unroll
    for (int d2 = 0; d2 < 16; ++d2) acc[d2] = 0.f;
    for (int j = 0; j < 49; ++j) {
      if (maskL[j] != 0.f) {
        const h16* kk = &qkvL[j][64 + h * 16];
        float s = 0.f;
#pragma unroll
        for (int d2 = 0; d2 < 16; ++d2) s += (float)q[d2] * (float)kk[d2];
        float wgt = __expf(s * 0.25f - mx);
        sum += wgt;
        const h16* vv = &qkvL[j][128 + h * 16];
#pragma unroll
        for (int d2 = 0; d2 < 16; ++d2) acc[d2] += wgt * (float)vv[d2];
      }
    }
    float inv = 1.f / sum;
#pragma unroll
    for (int d2 = 0; d2 < 16; ++d2) attnL[e][h * 16 + d2] = (h16)(acc[d2] * inv);
  }
  __syncthreads();

  // res = z_ent + relu(attn @ Wcomb.T + bcomb); pool += res*mask/49
#pragma unroll
  for (int i = 0; i < 2; ++i) {
    int t = w * 2 + i, rt = t & 3, ct = t >> 2;
    v8f d = wmma_kloop(&attnL[rt * 16][0], 64, wcomb + ct * 16 * EF, EF, EF);
    int n = ct * 16 + (lane & 15), e0 = rt * 16 + ((lane >> 4) << 3);
    float b = bcomb[n];
#pragma unroll
    for (int r = 0; r < 8; ++r) {
      int e = e0 + r;
      float res = (float)zL[e * EF + n] + fmaxf(d[r] + b, 0.f);
      if (e < 49) atomicAdd(&zpoolL[n], res * maskL[e] * (1.f / 49.f));
    }
  }
  __syncthreads();
  if (tid < 64) xpool[(size_t)m * XROW + SPAD + tid] = (h16)zpoolL[tid];
}

__global__ __launch_bounds__(256) void gemm_pool_kernel(
    const h16* A, const h16* W, const float* bias, h16* Cz) {
  __shared__ __align__(16) h16 Bs[16 * 384];
  int w = threadIdx.x >> 5, lane = threadIdx.x & 31;
  int row0 = (blockIdx.y * 8 + w) * 16, col0 = blockIdx.x * 16;
  stage_rows(W + (size_t)col0 * XROW, XROW, 16, 384, Bs);
  v8f d = wmma_kloop(A + (size_t)row0 * XROW, XROW, Bs, 384, 384);
  int n = col0 + (lane & 15), mb = row0 + ((lane >> 4) << 3);
  float b = bias[n];
#pragma unroll
  for (int r = 0; r < 8; ++r)
    Cz[(size_t)(mb + r) * 128 + n] = (h16)fmaxf(d[r] + b, 0.f);
}

__global__ __launch_bounds__(256) void gemm_lstmx_kernel(
    const h16* A, const h16* W, const float* b_ih, const float* b_hh, float* C) {
  __shared__ __align__(16) h16 Bs[16 * 128];
  int w = threadIdx.x >> 5, lane = threadIdx.x & 31;
  int row0 = (blockIdx.y * 8 + w) * 16, col0 = blockIdx.x * 16;
  stage_rows(W + (size_t)col0 * 128, 128, 16, 128, Bs);
  v8f d = wmma_kloop(A + (size_t)row0 * 128, 128, Bs, 128, 128);
  int n = col0 + (lane & 15), mb = row0 + ((lane >> 4) << 3);
  float b = b_ih[n] + b_hh[n];
#pragma unroll
  for (int r = 0; r < 8; ++r) C[(size_t)(mb + r) * 512 + n] = d[r] + b;
}

// one LSTM timestep: keep*(h@w_hh.T) + gates_x -> (h,c), value head
__global__ __launch_bounds__(256) void lstm_step_kernel(
    const float* gates_x, const h16* whh, h16* hbuf, float* cbuf,
    const float* done, const float* head_w, const float* head_b,
    float* out, int t) {
  __shared__ __align__(16) h16 hL[16 * 128];
  __shared__ __align__(16) float gl[16][512];
  __shared__ float hl[16][128];
  int tid = threadIdx.x, w = tid >> 5, lane = tid & 31;
  int r0 = blockIdx.x * 16;
  stage_rows(hbuf + (size_t)r0 * 128, 128, 16, 128, hL);
#pragma unroll
  for (int i = 0; i < 4; ++i) {
    int ct = w * 4 + i;
    v8f d = wmma_kloop(hL, 128, whh + (size_t)ct * 16 * 128, 128, 128);
    int n = ct * 16 + (lane & 15), mb = (lane >> 4) << 3;
#pragma unroll
    for (int r = 0; r < 8; ++r) gl[mb + r][n] = d[r];
  }
  __syncthreads();
  for (int u = tid; u < 16 * 128; u += 256) {
    int row = u >> 7, n = u & 127, gr = r0 + row;
    float keep = 1.f - done[t * 16 + (gr >> 3)];
    const float* gx = gates_x + ((size_t)t * 128 + gr) * 512;
    float ig = gx[n]       + keep * gl[row][n];
    float fg = gx[128 + n] + keep * gl[row][128 + n];
    float gg = gx[256 + n] + keep * gl[row][256 + n];
    float og = gx[384 + n] + keep * gl[row][384 + n];
    float c = cbuf[gr * 128 + n] * keep;
    c = sigm(fg) * c + sigm(ig) * tanhf(gg);
    float h = sigm(og) * tanhf(c);
    cbuf[gr * 128 + n] = c;
    hbuf[gr * 128 + n] = (h16)h;
    hl[row][n] = h;
  }
  __syncthreads();
  if (tid < 16) {
    float acc = head_b[0];
    for (int k = 0; k < 128; ++k) acc += hl[tid][k] * head_w[k];
    out[t * 128 + r0 + tid] = acc;
  }
}

// ---------------------------------------------------------------------------
extern "C" void kernel_launch(void* const* d_in, const int* in_sizes, int n_in,
                              void* d_out, int out_size, void* d_ws, size_t ws_size,
                              hipStream_t stream) {
  (void)in_sizes; (void)n_in; (void)out_size; (void)ws_size;
  const float* x_agent  = (const float*)d_in[0];
  const float* x_lidar  = (const float*)d_in[1];
  const float* x_safe   = (const float*)d_in[2];
  const float* feats0   = (const float*)d_in[3];
  const float* mask0    = (const float*)d_in[4];
  const float* feats1   = (const float*)d_in[5];
  const float* mask1    = (const float*)d_in[6];
  const float* feats2   = (const float*)d_in[7];
  const float* mask2    = (const float*)d_in[8];
  const float* done     = (const float*)d_in[9];
  const float* h0       = (const float*)d_in[10];
  const float* c0       = (const float*)d_in[11];
  const float* lidar_w  = (const float*)d_in[12];
  const float* lidar_b  = (const float*)d_in[13];
  const float* self_w   = (const float*)d_in[14];
  const float* self_b   = (const float*)d_in[15];
  const float* ent_w0   = (const float*)d_in[16];
  const float* ent_b0   = (const float*)d_in[17];
  const float* ent_w1   = (const float*)d_in[18];
  const float* ent_b1   = (const float*)d_in[19];
  const float* ent_w2   = (const float*)d_in[20];
  const float* ent_b2   = (const float*)d_in[21];
  const float* in_proj_w = (const float*)d_in[22];
  const float* in_proj_b = (const float*)d_in[23];
  const float* out_proj_w = (const float*)d_in[24];
  const float* out_proj_b = (const float*)d_in[25];
  const float* sa_w     = (const float*)d_in[26];
  const float* sa_b     = (const float*)d_in[27];
  const float* pool_w   = (const float*)d_in[28];
  const float* pool_b   = (const float*)d_in[29];
  const float* w_ih     = (const float*)d_in[30];
  const float* w_hh     = (const float*)d_in[31];
  const float* b_ih     = (const float*)d_in[32];
  const float* b_hh     = (const float*)d_in[33];
  const float* head_w   = (const float*)d_in[34];
  const float* head_b   = (const float*)d_in[35];
  float* out = (float*)d_out;

  char* p = (char*)d_ws;
  auto take = [&](size_t n) { char* q = p; p += (n + 255) & ~(size_t)255; return q; };
  h16*   xpool   = (h16*)  take((size_t)MTOK * XROW * 2);
  h16*   Wc      = (h16*)  take(256 * 320 * 2);
  float* g0      = (float*)take((size_t)MTOK * 256 * 4);
  h16*   featpad = (h16*)  take((size_t)3 * 65536 * 32 * 2);
  h16*   wfeat   = (h16*)  take(3 * 64 * 32 * 2);
  h16*   zent    = (h16*)  take((size_t)MTOK * EPAD * EF * 2);
  h16*   inproj  = (h16*)  take(192 * 64 * 2);
  h16*   wcomb   = (h16*)  take(64 * 64 * 2);
  float* bcomb   = (float*)take(64 * 4);
  h16*   poolw   = (h16*)  take(128 * 384 * 2);
  h16*   zbuf    = (h16*)  take((size_t)MTOK * 128 * 2);
  h16*   wih     = (h16*)  take(512 * 128 * 2);
  h16*   whh     = (h16*)  take(512 * 128 * 2);
  float* gx      = (float*)take((size_t)MTOK * 512 * 4);
  h16*   hbuf    = (h16*)  take(128 * 128 * 2);
  float* cbuf    = (float*)take(128 * 128 * 4);

  prep_misc_kernel<<<1160, 256, 0, stream>>>(
      self_w, ent_w0, ent_w1, ent_w2, in_proj_w, pool_w, w_ih, w_hh, h0, c0,
      Wc, wfeat, inproj, poolw, wih, whh, hbuf, cbuf);
  prep_comb_kernel<<<16, 256, 0, stream>>>(sa_w, sa_b, out_proj_w, out_proj_b,
                                           wcomb, bcomb);
  build_xself_kernel<<<MTOK, 64, 0, stream>>>(x_agent, x_lidar, x_safe,
                                              lidar_w, lidar_b, xpool);
  featpad_kernel<<<24576, 256, 0, stream>>>(feats0, feats1, feats2, featpad);

  gemm_g0_kernel<<<dim3(16, 32), 256, 0, stream>>>(xpool, Wc, g0);
  zself_kernel<<<16384, 256, 0, stream>>>(g0, self_b, zent);

  gemm_feat_kernel<<<dim3(4, 512), 256, 0, stream>>>(featpad, wfeat,
      g0, 64, ent_b0, zent, 1);
  gemm_feat_kernel<<<dim3(4, 512), 256, 0, stream>>>(featpad + (size_t)65536 * 32,
      wfeat + 2048, g0, 128, ent_b1, zent, 17);
  gemm_feat_kernel<<<dim3(4, 512), 256, 0, stream>>>(featpad + (size_t)2 * 65536 * 32,
      wfeat + 4096, g0, 192, ent_b2, zent, 33);

  attn_kernel<<<MTOK, 256, 0, stream>>>(zent, inproj, in_proj_b, wcomb, bcomb,
                                        mask0, mask1, mask2, xpool);

  gemm_pool_kernel<<<dim3(8, 32), 256, 0, stream>>>(xpool, poolw, pool_b, zbuf);
  gemm_lstmx_kernel<<<dim3(32, 32), 256, 0, stream>>>(zbuf, wih, b_ih, b_hh, gx);

  for (int t = 0; t < 32; ++t)
    lstm_step_kernel<<<8, 256, 0, stream>>>(gx, whh, hbuf, cbuf, done,
                                            head_w, head_b, out, t);
}
